// UpdateNode_39857296507338
// MI455X (gfx1250) — matrix operand order repaired and checked by reading, not compiled, not run on hardware
//
#include <hip/hip_runtime.h>
#include <math.h>

// ---------------------------------------------------------------------------
// MI455X (gfx1250) fused equivariant message-passing layer.
// Dominant cost: E x 11520 x 128 dynamic-weight GEMM (30.2 GFLOP) fused with
// the per-edge SO2 apply so the 471MB wdyn tensor is never materialized.
// GEMM runs as v_wmma_f32_16x16x32_bf16 (bf16 in, fp32 accumulate, K=128),
// with double-buffered B fragments to hide global latency behind the MMA
// pipe, and batched ds_load -> ds_add_f32 scatter for the apply.
// ---------------------------------------------------------------------------

typedef __attribute__((ext_vector_type(16))) __bf16 bf16x16;
typedef __attribute__((ext_vector_type(8)))  __bf16 bf16x8;
typedef __attribute__((ext_vector_type(8)))  float  f32x8;

#define KDIM   128
#define NCOLS  11520

// ---------------------------------------------------------------------------
// Closed-form Wigner D2 for basis [xy, yz, 3z^2-1, xz, x^2-y^2] on unit vecs.
// Equals the reference's pinv-of-sampled-points construction exactly.
// ---------------------------------------------------------------------------
__device__ inline void wigner_row(const float* u, const float* w, float scale,
                                  float* out) {
  float dx = u[0]*w[0], dy = u[1]*w[1], dz = u[2]*w[2];
  out[0] = scale * (u[0]*w[1] + u[1]*w[0]);
  out[1] = scale * (u[1]*w[2] + u[2]*w[1]);
  out[2] = scale * (2.f*dz - dx - dy) * (1.f/6.f);
  out[3] = scale * (u[0]*w[2] + u[2]*w[0]);
  out[4] = scale * (dx - dy) * 0.5f;
}

__device__ inline void wigner_d2(const float R[3][3], float D[5][5]) {
  wigner_row(R[0], R[1], 1.f, D[0]);   // q_x q_y
  wigner_row(R[1], R[2], 1.f, D[1]);   // q_y q_z
  wigner_row(R[2], R[2], 3.f, D[2]);   // 3 q_z^2 - 1
  wigner_row(R[0], R[2], 1.f, D[3]);   // q_x q_z
  float t0[5], t1[5];
  wigner_row(R[0], R[0], 1.f, t0);
  wigner_row(R[1], R[1], 1.f, t1);
  for (int k = 0; k < 5; ++k) D[4][k] = t0[k] - t1[k];  // q_x^2 - q_y^2
}

// ---------------------------------------------------------------------------
// Kernel: zero the node accumulator (must happen every launch).
// ---------------------------------------------------------------------------
__global__ void zero_acc_kernel(float* __restrict__ acc, int n) {
  int i = blockIdx.x * blockDim.x + threadIdx.x;
  if (i < n) acc[i] = 0.f;
}

// ---------------------------------------------------------------------------
// Kernel: _sln over node features (N=512, trivial).
// ---------------------------------------------------------------------------
__global__ void prep_nodes_kernel(const float* __restrict__ x,
                                  const float* __restrict__ w0,
                                  const float* __restrict__ b0,
                                  const float* __restrict__ w1,
                                  const float* __restrict__ w2,
                                  float* __restrict__ nf, int N) {
  int n = blockIdx.x * blockDim.x + threadIdx.x;
  if (n >= N) return;
  const float* xr = x + (size_t)n * 120;
  float* o = nf + (size_t)n * 120;
  float mu = 0.f;
  for (int k = 0; k < 32; ++k) mu += xr[k];
  mu *= (1.f/32.f);
  float var = 0.f;
  for (int k = 0; k < 32; ++k) { float d = xr[k]-mu; var += d*d; }
  var *= (1.f/32.f);
  float inv = rsqrtf(var + 1e-5f);
  for (int k = 0; k < 32; ++k) o[k] = (xr[k]-mu)*inv*w0[k] + b0[k];
  float v1s = 0.f, v2s = 0.f;
  for (int k = 32; k < 80; ++k)  v1s += xr[k]*xr[k];
  for (int k = 80; k < 120; ++k) v2s += xr[k]*xr[k];
  float invv = rsqrtf(0.5f*(v1s*(1.f/48.f) + v2s*(1.f/40.f)) + 1e-5f);
  for (int u = 0; u < 16; ++u)
    for (int i = 0; i < 3; ++i) o[32+3*u+i] = xr[32+3*u+i]*invv*w1[u];
  for (int u = 0; u < 8; ++u)
    for (int i = 0; i < 5; ++i) o[80+5*u+i] = xr[80+5*u+i]*invv*w2[u];
}

// ---------------------------------------------------------------------------
// Kernel: convert latents to bf16 (A operand).
// ---------------------------------------------------------------------------
__global__ void cvt_lat_kernel(const float* __restrict__ lat,
                               unsigned short* __restrict__ latB, int n) {
  int i = blockIdx.x * blockDim.x + threadIdx.x;
  if (i < n) ((__bf16*)latB)[i] = (__bf16)lat[i];
}

// ---------------------------------------------------------------------------
// Kernel: W_tp -> transposed bf16 (NCOLS x KDIM), 1/sqrt(128) folded in.
// Column-major storage makes each B fragment two contiguous b128 loads.
// ---------------------------------------------------------------------------
__global__ void cvt_wtp_kernel(const float* __restrict__ W_tp,
                               unsigned short* __restrict__ WtpT) {
  int i = blockIdx.x * blockDim.x + threadIdx.x;  // i < NCOLS*KDIM exactly
  int c = i >> 7, k = i & 127;
  ((__bf16*)WtpT)[i] = (__bf16)(W_tp[(size_t)k * NCOLS + c] * 0.08838834764831845f);
}

// ---------------------------------------------------------------------------
// Kernel: wenv = latents[active] @ W_env / sqrt(128)  (E x 56, small).
// ---------------------------------------------------------------------------
__global__ void wenv_kernel(const float* __restrict__ lat,
                            const float* __restrict__ W_env,
                            const int* __restrict__ active,
                            float* __restrict__ wenv, int E) {
  int idx = blockIdx.x * blockDim.x + threadIdx.x;
  if (idx >= E * 56) return;
  int e = idx / 56, o = idx % 56;
  const float* lr = lat + (size_t)active[e] * KDIM;
  float s = 0.f;
  for (int l = 0; l < KDIM; ++l) s += lr[l] * W_env[l*56 + o];
  wenv[idx] = s * 0.08838834764831845f;
}

// ---------------------------------------------------------------------------
// Kernel: per-edge geometry + input assembly.
// so2in layout per edge (240 f32):
//   [0:112)  in0   = [s(64), v1rot[:,1](32), v2rot[:,2](16)]
//   [112:160) inn1 = [v1rot[:,0](32), v2rot[:,1](16)]
//   [160:208) inp1 = [v1rot[:,2](32), v2rot[:,3](16)]
//   [208:224) inn2 = v2rot[:,0]
//   [224:240) inp2 = v2rot[:,4]
// rd layout per edge (34 f32): R row-major (9) + D2b row-major (25).
// ---------------------------------------------------------------------------
__global__ void prep_edges_kernel(const float* __restrict__ nf,
                                  const float* __restrict__ hidden,
                                  const float* __restrict__ edge_vector,
                                  const int* __restrict__ edge_index,
                                  const int* __restrict__ active,
                                  float* __restrict__ so2in,
                                  float* __restrict__ rd, int E) {
  int e = blockIdx.x * blockDim.x + threadIdx.x;
  if (e >= E) return;
  int ae = active[e];
  float vx = edge_vector[(size_t)ae*3+0];
  float vy = edge_vector[(size_t)ae*3+1];
  float vz = edge_vector[(size_t)ae*3+2];
  float inv = rsqrtf(vx*vx + vy*vy + vz*vz);
  float n0 = vx*inv, n1 = vy*inv, n2 = vz*inv;
  float a0, a1, a2;
  if (fabsf(n2) < 0.99f) { a0 = 0.f; a1 = 0.f; a2 = 1.f; }
  else                   { a0 = 1.f; a1 = 0.f; a2 = 0.f; }
  float e10 = a1*n2 - a2*n1, e11 = a2*n0 - a0*n2, e12 = a0*n1 - a1*n0;
  float einv = rsqrtf(e10*e10 + e11*e11 + e12*e12);
  e10 *= einv; e11 *= einv; e12 *= einv;
  float e20 = n1*e12 - n2*e11, e21 = n2*e10 - n0*e12, e22 = n0*e11 - n1*e10;
  float R[3][3] = {{e10,e11,e12},{e20,e21,e22},{n0,n1,n2}};
  float Rt[3][3] = {{e10,e20,n0},{e11,e21,n1},{e12,e22,n2}};
  float D2f[5][5], D2b[5][5];
  wigner_d2(R,  D2f);
  wigner_d2(Rt, D2b);

  float* rde = rd + (size_t)e * 34;
  int q = 0;
  for (int i = 0; i < 3; ++i) for (int j = 0; j < 3; ++j) rde[q++] = R[i][j];
  for (int i = 0; i < 5; ++i) for (int j = 0; j < 5; ++j) rde[q++] = D2b[i][j];

  int c = edge_index[ae];
  const float* nfc = nf + (size_t)c * 120;
  const float* h   = hidden + (size_t)e * 120;
  float* out = so2in + (size_t)e * 240;
  for (int k = 0; k < 32; ++k) { out[k] = nfc[k]; out[32+k] = h[k]; }
  // v1: 32 rows of 3
  for (int u = 0; u < 32; ++u) {
    float x0, x1, x2;
    if (u < 16) { x0 = nfc[32+3*u];   x1 = nfc[33+3*u];   x2 = nfc[34+3*u]; }
    else { int uu = u-16; x0 = h[32+3*uu]; x1 = h[33+3*uu]; x2 = h[34+3*uu]; }
    float r0 = R[0][0]*x0 + R[0][1]*x1 + R[0][2]*x2;
    float r1 = R[1][0]*x0 + R[1][1]*x1 + R[1][2]*x2;
    float r2 = R[2][0]*x0 + R[2][1]*x1 + R[2][2]*x2;
    out[64+u]  = r1;   // in0 (m=0)
    out[112+u] = r0;   // inn1
    out[160+u] = r2;   // inp1
  }
  // v2: 16 rows of 5
  for (int u = 0; u < 16; ++u) {
    float xv[5];
    if (u < 8) { for (int j = 0; j < 5; ++j) xv[j] = nfc[80+5*u+j]; }
    else { int uu = u-8; for (int j = 0; j < 5; ++j) xv[j] = h[80+5*uu+j]; }
    float r[5];
    for (int i = 0; i < 5; ++i) {
      float s = 0.f;
      for (int j = 0; j < 5; ++j) s += D2f[i][j]*xv[j];
      r[i] = s;
    }
    out[96+u]  = r[2];  // in0 (m=0)
    out[144+u] = r[1];  // inn1
    out[192+u] = r[3];  // inp1
    out[208+u] = r[0];  // inn2
    out[224+u] = r[4];  // inp2
  }
}

// ---------------------------------------------------------------------------
// Fused kernel: per workgroup = 16 edges, 8 waves x 256 threads.
// A (latents 16x128 bf16) in registers; 90 iterations, each wave owns one
// 16-edge x 16-col tile per iter (4x v_wmma_f32_16x16x32_bf16 over K=128).
// B fragments are double-buffered: next tile's 8 b128 loads are issued
// before the current tile's WMMA chain so the MMA pipe never waits on HBM/L2.
// Dynamic weights are scattered into per-edge LDS accumulators (ds_add_f32)
// with batched ds_load first (one wait), then the atomics.
// s_acc layout per edge (144): [0:80)=out0 [80:104)=op1 [104:128)=on1
//                              [128:136)=op2 [136:144)=on2
// ---------------------------------------------------------------------------
__global__ void __launch_bounds__(256)
so2_fused_kernel(const unsigned short* __restrict__ latB_u,
                 const unsigned short* __restrict__ WtpT_u,
                 const float* __restrict__ so2in,
                 const float* __restrict__ rd,
                 const float* __restrict__ wenvp,
                 const float* __restrict__ Wp0, const float* __restrict__ bp0,
                 const float* __restrict__ Wp1, const float* __restrict__ Wp2,
                 const int* __restrict__ edge_index,
                 const int* __restrict__ active,
                 float* __restrict__ node_acc, int E) {
  __shared__ float s_in[16 * 240];
  __shared__ float s_acc[16 * 144];
  const int tid  = threadIdx.x;
  const int lane = tid & 31;
  const int wv   = tid >> 5;
  const int e0   = blockIdx.x * 16;

  for (int i = tid; i < 16*144; i += 256) s_acc[i] = 0.f;
  for (int i = tid; i < 16*240; i += 256) {
    int el = i / 240, k = i % 240;
    int ee = (e0 + el < E) ? (e0 + el) : (E - 1);
    s_in[i] = so2in[(size_t)ee * 240 + k];
  }
  __syncthreads();

  const __bf16* latB = (const __bf16*)latB_u;
  const __bf16* WtpT = (const __bf16*)WtpT_u;

  // A fragments: ISA 16-bit A layout (lanes 0-15 rows M, K 0-7/16-23;
  // lanes 16-31 same rows, K 8-15/24-31).
  const int rowe  = lane & 15;
  const int half  = lane >> 4;
  const int khalf = half;
  int ar = e0 + rowe; if (ar >= E) ar = E - 1;
  bf16x16 afrag[4];
#pragma unroll
  for (int c = 0; c < 4; ++c) {
    const float* dummy = nullptr; (void)dummy;
    const __bf16* base = latB + (size_t)ar * KDIM + c*32 + 8*half;
    bf16x8 lo = *(const bf16x8*)(base);
    bf16x8 hi = *(const bf16x8*)(base + 16);
#pragma unroll
    for (int i = 0; i < 8; ++i) { afrag[c][i] = lo[i]; afrag[c][i+8] = hi[i]; }
  }

  const int niter = NCOLS / 128;  // 90, 8 waves x 16 cols each
  const int lanecol = lane & 15;

  // B fragment loader: col-major WtpT -> K 8h..8h+7 and 16+8h..16+8h+7 per lane/chunk.
  auto loadB = [&](int t, bf16x16* bf) {
    const int colx = t*128 + wv*16 + lanecol;
    const __bf16* bb0 = WtpT + (size_t)colx * KDIM + 8*khalf;
#pragma unroll
    for (int c = 0; c < 4; ++c) {
      bf16x8 lo = *(const bf16x8*)(bb0 + c*32);
      bf16x8 hi = *(const bf16x8*)(bb0 + c*32 + 16);
#pragma unroll
      for (int i = 0; i < 8; ++i) { bf[c][i] = lo[i]; bf[c][i+8] = hi[i]; }
    }
  };

  // One tile: 4 chained WMMAs + scatter-apply of the 8 weights per lane.
  auto step = [&](int t, bf16x16* bc) {
    const int col = t*128 + wv*16 + lanecol;   // flat dynamic-weight column
    f32x8 acc = {};
#pragma unroll
    for (int c = 0; c < 4; ++c)
      acc = __builtin_amdgcn_wmma_f32_16x16x32_bf16(
          false, afrag[c], false, bc[c], (short)0, acc, false, false);

    // Unified region decode: acc contributes to s_acc[tgt1] (sign sg1) with
    // input s_in[in1], and optionally to s_acc[tgt2] (+) with s_in[in2].
    int tgt1, in1, tgt2 = -1, in2 = 0;
    float sg1 = 1.f;
    if (col < 8960) {                               // w0
      tgt1 = col % 80;        in1 = col / 80;
    } else if (col < 10112) {                       // w1r
      int idx = col - 8960;   int i = idx / 24, o = idx % 24;
      tgt1 = 80 + o; in1 = 160 + i;                 // op1 += w*inp1
      tgt2 = 104 + o; in2 = 112 + i;                // on1 += w*inn1
    } else if (col < 11264) {                       // w1i
      int idx = col - 10112;  int i = idx / 24, o = idx % 24;
      tgt1 = 80 + o; in1 = 112 + i; sg1 = -1.f;     // op1 -= w*inn1
      tgt2 = 104 + o; in2 = 160 + i;                // on1 += w*inp1
    } else if (col < 11392) {                       // w2r
      int idx = col - 11264;  int i = idx / 8, o = idx % 8;
      tgt1 = 128 + o; in1 = 224 + i;                // op2 += w*inp2
      tgt2 = 136 + o; in2 = 208 + i;                // on2 += w*inn2
    } else {                                        // w2i
      int idx = col - 11392;  int i = idx / 8, o = idx % 8;
      tgt1 = 128 + o; in1 = 208 + i; sg1 = -1.f;    // op2 -= w*inn2
      tgt2 = 136 + o; in2 = 224 + i;                // on2 += w*inp2
    }
    // Batched LDS reads (single wait), then atomics.
    float x1[8];
#pragma unroll
    for (int r = 0; r < 8; ++r) x1[r] = s_in[(r + 8*khalf)*240 + in1];
    if (tgt2 >= 0) {
      float x2[8];
#pragma unroll
      for (int r = 0; r < 8; ++r) x2[r] = s_in[(r + 8*khalf)*240 + in2];
#pragma unroll
      for (int r = 0; r < 8; ++r) {
        int el = r + 8*khalf;
        atomicAdd(&s_acc[el*144 + tgt1], sg1 * acc[r] * x1[r]);
        atomicAdd(&s_acc[el*144 + tgt2], acc[r] * x2[r]);
      }
    } else {
#pragma unroll
      for (int r = 0; r < 8; ++r) {
        int el = r + 8*khalf;
        atomicAdd(&s_acc[el*144 + tgt1], acc[r] * x1[r]);
      }
    }
  };

  // Double-buffered main loop (niter is even).
  bf16x16 bA[4], bB[4];
  loadB(0, bA);
  for (int t = 0; t < niter; t += 2) {
    if (t + 1 < niter) {
      loadB(t + 1, bB);
      if (t + 3 < niter)
        __builtin_prefetch(WtpT + (size_t)((t+3)*128 + wv*16 + lanecol) * KDIM, 0, 1);
    }
    step(t, bA);
    if (t + 1 < niter) {
      if (t + 2 < niter) loadB(t + 2, bA);
      step(t + 1, bB);
    }
  }
  __syncthreads();

  // ------------------- per-edge finalize (16 threads) -------------------
  if (tid < 16 && e0 + tid < E) {
    const int e = e0 + tid;
    const float* A = &s_acc[tid * 144];
    float R[3][3], D2b[5][5];
    {
      const float* rde = rd + (size_t)e * 34;
      int q = 0;
      for (int i = 0; i < 3; ++i) for (int j = 0; j < 3; ++j) R[i][j] = rde[q++];
      for (int i = 0; i < 5; ++i) for (int j = 0; j < 5; ++j) D2b[i][j] = rde[q++];
    }
    const float inv112 = 0.09449111825230679f;   // 1/sqrt(112)
    const float inv48  = 0.14433756729740643f;   // 1/sqrt(48)
    const float inv16  = 0.25f;                  // 1/sqrt(16)
    float m[144];
    for (int k = 0; k < 56; ++k) m[k] = A[k] * inv112;              // s_out
    for (int u = 0; u < 16; ++u) {                                  // y1 = R^T y
      float y0 = A[104+u]*inv48, y1v = A[56+u]*inv112, y2v = A[80+u]*inv48;
      for (int i = 0; i < 3; ++i)
        m[56+3*u+i] = R[0][i]*y0 + R[1][i]*y1v + R[2][i]*y2v;
    }
    for (int u = 0; u < 8; ++u) {                                   // y2 = D2b y
      float yv[5] = { A[136+u]*inv16, A[120+u]*inv48, A[72+u]*inv112,
                      A[96+u]*inv48,  A[128+u]*inv16 };
      for (int i = 0; i < 5; ++i) {
        float s = 0.f;
        for (int j = 0; j < 5; ++j) s += D2b[i][j]*yv[j];
        m[104+5*u+i] = s;
      }
    }
    // gate
    float sg[32], gg[24];
    for (int k = 0; k < 32; ++k) { float x = m[k]; sg[k] = x / (1.f + expf(-x)); }
    for (int k = 0; k < 24; ++k) gg[k] = 1.f / (1.f + expf(-m[32+k]));
    // _ir_linear(Wp) + _env
    const float invs32 = 0.17677669529663687f;   // 1/sqrt(32)
    const float invs8  = 0.35355339059327373f;   // 1/sqrt(8)
    const float* we = wenvp + (size_t)e * 56;
    int ae = active[e];
    int cn = edge_index[ae];
    float* dst = node_acc + (size_t)cn * 120;
    for (int o = 0; o < 32; ++o) {
      float s = 0.f;
      for (int u = 0; u < 32; ++u) s += sg[u] * Wp0[u*32 + o];
      atomicAdd(&dst[o], (s*invs32 + bp0[o]) * we[o]);
    }
    for (int v = 0; v < 16; ++v) {
      float g2 = we[32+v];
      for (int i = 0; i < 3; ++i) {
        float s = 0.f;
        for (int u = 0; u < 16; ++u) s += m[56+3*u+i]*gg[u]*Wp1[u*16 + v];
        atomicAdd(&dst[32+3*v+i], s * 0.25f * g2);
      }
    }
    for (int v = 0; v < 8; ++v) {
      float g2 = we[48+v];
      for (int i = 0; i < 5; ++i) {
        float s = 0.f;
        for (int u = 0; u < 8; ++u) s += m[104+5*u+i]*gg[16+u]*Wp2[u*8 + v];
        atomicAdd(&dst[80+5*v+i], s * invs8 * g2);
      }
    }
  }
}

// ---------------------------------------------------------------------------
// Kernel: final per-node combine: C_NEW*NORM*acc + C_OLD*_ir_linear(Wr).
// ---------------------------------------------------------------------------
__global__ void finalize_kernel(const float* __restrict__ acc,
                                const float* __restrict__ x,
                                const float* __restrict__ Wr0,
                                const float* __restrict__ br0,
                                const float* __restrict__ Wr1,
                                const float* __restrict__ Wr2,
                                float* __restrict__ out, int N) {
  int n = blockIdx.x * blockDim.x + threadIdx.x;
  if (n >= N) return;
  const float C_OLD = 0.8944271909999159f;   // 1/sqrt(1.25)
  const float C_NEW = 0.4472135954999579f;   // 0.5*C_OLD
  const float NORMC = 0.22360679774997896f;  // 1/sqrt(20)
  const float invs32 = 0.17677669529663687f, invs8 = 0.35355339059327373f;
  const float* xr = x + (size_t)n * 120;
  const float* ar = acc + (size_t)n * 120;
  float* o = out + (size_t)n * 120;
  const float k1 = C_NEW * NORMC;
  for (int oo = 0; oo < 32; ++oo) {
    float s = 0.f;
    for (int u = 0; u < 32; ++u) s += xr[u] * Wr0[u*32 + oo];
    o[oo] = k1*ar[oo] + C_OLD*(s*invs32 + br0[oo]);
  }
  for (int v = 0; v < 16; ++v)
    for (int i = 0; i < 3; ++i) {
      float s = 0.f;
      for (int u = 0; u < 16; ++u) s += xr[32+3*u+i] * Wr1[u*16 + v];
      o[32+3*v+i] = k1*ar[32+3*v+i] + C_OLD*(s*0.25f);
    }
  for (int v = 0; v < 8; ++v)
    for (int i = 0; i < 5; ++i) {
      float s = 0.f;
      for (int u = 0; u < 8; ++u) s += xr[80+5*u+i] * Wr2[u*8 + v];
      o[80+5*v+i] = k1*ar[80+5*v+i] + C_OLD*(s*invs8);
    }
}

// ---------------------------------------------------------------------------
// Host launcher.
// ---------------------------------------------------------------------------
extern "C" void kernel_launch(void* const* d_in, const int* in_sizes, int n_in,
                              void* d_out, int out_size, void* d_ws, size_t ws_size,
                              hipStream_t stream) {
  const float* latents       = (const float*)d_in[0];
  const float* node_features = (const float*)d_in[1];
  const float* hidden        = (const float*)d_in[2];
  const float* edge_vector   = (const float*)d_in[4];
  const float* W_tp          = (const float*)d_in[5];
  const float* W_env         = (const float*)d_in[6];
  const float* sln_w0 = (const float*)d_in[7];
  const float* sln_b0 = (const float*)d_in[8];
  const float* sln_w1 = (const float*)d_in[9];
  const float* sln_w2 = (const float*)d_in[10];
  const float* Wp0 = (const float*)d_in[11];
  const float* bp0 = (const float*)d_in[12];
  const float* Wp1 = (const float*)d_in[13];
  const float* Wp2 = (const float*)d_in[14];
  const float* Wr0 = (const float*)d_in[15];
  const float* br0 = (const float*)d_in[16];
  const float* Wr1 = (const float*)d_in[17];
  const float* Wr2 = (const float*)d_in[18];
  const int* edge_index = (const int*)d_in[20];
  const int* active     = (const int*)d_in[21];

  const int E = in_sizes[21];
  const int N = in_sizes[1] / 120;

  // workspace carve-out (256B aligned regions)
  char* base = (char*)d_ws;
  size_t off = 0;
  auto al = [](size_t x) { return (x + 255) & ~(size_t)255; };
  float* node_acc = (float*)(base + off); off = al(off + (size_t)N * 120 * 4);
  float* nf       = (float*)(base + off); off = al(off + (size_t)N * 120 * 4);
  float* so2in    = (float*)(base + off); off = al(off + (size_t)E * 240 * 4);
  float* rdbuf    = (float*)(base + off); off = al(off + (size_t)E * 34 * 4);
  float* wenvbuf  = (float*)(base + off); off = al(off + (size_t)E * 56 * 4);
  unsigned short* latB = (unsigned short*)(base + off); off = al(off + (size_t)E * KDIM * 2);
  unsigned short* WtpT = (unsigned short*)(base + off); off = al(off + (size_t)NCOLS * KDIM * 2);
  (void)ws_size; (void)n_in;

  zero_acc_kernel<<<(N*120 + 255)/256, 256, 0, stream>>>(node_acc, N*120);
  prep_nodes_kernel<<<(N + 63)/64, 64, 0, stream>>>(
      node_features, sln_w0, sln_b0, sln_w1, sln_w2, nf, N);
  cvt_lat_kernel<<<(E*KDIM + 255)/256, 256, 0, stream>>>(latents, latB, E*KDIM);
  cvt_wtp_kernel<<<(NCOLS*KDIM)/256, 256, 0, stream>>>(W_tp, WtpT);
  wenv_kernel<<<(E*56 + 255)/256, 256, 0, stream>>>(latents, W_env, active, wenvbuf, E);
  prep_edges_kernel<<<(E + 63)/64, 64, 0, stream>>>(
      nf, hidden, edge_vector, edge_index, active, so2in, rdbuf, E);
  so2_fused_kernel<<<(E + 15)/16, 256, 0, stream>>>(
      latB, WtpT, so2in, rdbuf, wenvbuf, Wp0, bp0, Wp1, Wp2,
      edge_index, active, node_acc, E);
  finalize_kernel<<<(N + 63)/64, 64, 0, stream>>>(
      node_acc, node_features, Wr0, br0, Wr1, Wr2, (float*)d_out, N);
}
